// ATAC_Encoder_87789131530807
// MI455X (gfx1250) — matrix operand order repaired and checked by reading, not compiled, not run on hardware
//
#include <hip/hip_runtime.h>
#include <hip/hip_bf16.h>

typedef unsigned short u16;
typedef __bf16 v16bf __attribute__((ext_vector_type(16)));
typedef float  v8f   __attribute__((ext_vector_type(8)));

#define LN_EPS 1e-5f
#define BN_EPS 1e-5f
#define SLOPE  0.01f

union FragBF { uint4 u[2]; v16bf v; };

__device__ __forceinline__ u16 f2b(float f) {
    unsigned u = __float_as_uint(f);
    unsigned r = (u + 0x7FFFu + ((u >> 16) & 1u)) >> 16;
    return (u16)r;
}

__device__ __forceinline__ float leaky(float x) { return x > 0.f ? x : SLOPE * x; }

// ---------------------------------------------------------------- convert f32 -> bf16 (8/thread)
__global__ void k_f32_to_bf16(const float* __restrict__ src, u16* __restrict__ dst, int n8) {
    int i = blockIdx.x * blockDim.x + threadIdx.x;
    if (i >= n8) return;
    const float4* s = (const float4*)(src) + 2 * (size_t)i;
    float4 a = s[0], b = s[1];
    uint4 o;
    o.x = (unsigned)f2b(a.x) | ((unsigned)f2b(a.y) << 16);
    o.y = (unsigned)f2b(a.z) | ((unsigned)f2b(a.w) << 16);
    o.z = (unsigned)f2b(b.x) | ((unsigned)f2b(b.y) << 16);
    o.w = (unsigned)f2b(b.z) | ((unsigned)f2b(b.w) << 16);
    *((uint4*)(dst) + (size_t)i) = o;
}

__global__ void k_zero_f32(float* p, int n) {
    int i = blockIdx.x * blockDim.x + threadIdx.x;
    if (i < n) p[i] = 0.f;
}

// ---------------------------------------------------------------- WMMA GEMM: C[N x COLS] = A[N x KD](bf16) * B[COLS x KD](bf16, row-major weights)
// block = WAVES waves; each wave: 16 rows x (TILES*16) cols; A tile staged in LDS
// via the gfx1250 async global->LDS path (no VGPR round trip).
template<int KD, int TILES, int WAVES>
__launch_bounds__(WAVES * 32)
__global__ void k_gemm_bf16_wmma(const u16* __restrict__ A,
                                 const u16* __restrict__ B,
                                 float* __restrict__ C) {
    constexpr int COLS = WAVES * TILES * 16;
    constexpr int LDSK = KD + 8;                 // pad 16B: conflict-free b128 reads
    __shared__ u16 sA[16 * LDSK];

    const int tid  = threadIdx.x;
    const int row0 = blockIdx.x * 16;

    // cooperative async stage of 16 x KD bf16 A tile: global -> LDS direct (ASYNCcnt)
    constexpr int VECS = 16 * KD / 8;            // 16-byte chunks
    const u16* Ablk = A + (size_t)row0 * KD;
    for (int i = tid; i < VECS; i += WAVES * 32) {
        int r = i / (KD / 8);
        int c = (i % (KD / 8)) * 8;
        const u16* gp = Ablk + (size_t)r * KD + c;
        unsigned lp = (unsigned)(uintptr_t)(&sA[r * LDSK + c]);   // LDS offset = addr[31:0]
        asm volatile("global_load_async_to_lds_b128 %0, %1, off"
                     :: "v"(lp), "v"((unsigned long long)(uintptr_t)gp)
                     : "memory");
    }
    asm volatile("s_wait_asynccnt 0" ::: "memory");
    __syncthreads();

    const int lane = tid & 31;
    const int wave = tid >> 5;
    const int m    = lane & 15;
    const int half = lane >> 4;

    v8f acc[TILES];
#pragma unroll
    for (int t = 0; t < TILES; ++t)
#pragma unroll
        for (int j = 0; j < 8; ++j) acc[t][j] = 0.f;

    const int n0 = wave * (TILES * 16);
    const u16* Bw = B + (size_t)(n0 + m) * KD;   // lane owns weight row n0+m (B col)

    for (int kb = 0; kb < KD; kb += 32) {
        __builtin_prefetch(Bw + kb + 64, 0, 3);  // speculative; OOB silently dropped
        // batch ALL fragment loads first ...
        FragBF b[TILES];
#pragma unroll
        for (int t = 0; t < TILES; ++t) {
            // B frag: elems 0..7 -> K = kb + half*8 + j ; elems 8..15 -> K = kb+16+half*8+j
            const u16* bp = Bw + (size_t)t * 16 * KD + kb + half * 8;
            b[t].u[0] = ((const uint4*)bp)[0];
            b[t].u[1] = ((const uint4*)(bp + 16))[0];
        }
        FragBF a;
        // A frag: elems 0..7 -> K = kb + half*8 + j ; elems 8..15 -> K = kb+16+half*8+j
        a.u[0] = *(const uint4*)(&sA[m * LDSK + kb + half * 8]);
        a.u[1] = *(const uint4*)(&sA[m * LDSK + kb + 16 + half * 8]);
        // ... then fence the scheduler so loads cannot be sunk between WMMAs:
        // one wait, then TILES back-to-back v_wmma ops with distinct live B buffers.
        __builtin_amdgcn_sched_barrier(0);
#pragma unroll
        for (int t = 0; t < TILES; ++t) {
            acc[t] = __builtin_amdgcn_wmma_f32_16x16x32_bf16(
                false, a.v, false, b[t].v, (short)0, acc[t], false, false);
        }
    }

    float* Crow = C + (size_t)row0 * COLS;
#pragma unroll
    for (int t = 0; t < TILES; ++t) {
        int nc = n0 + t * 16 + m;
#pragma unroll
        for (int r = 0; r < 8; ++r) {
            int mr = r + 8 * half;               // C/D layout: M = r + 8*(lane/16)
            Crow[(size_t)mr * COLS + nc] = acc[t][r];
        }
    }
}

// ---------------------------------------------------------------- BatchNorm column stats (cols 512..1023 of Y)
__global__ void k_bn_stats(const float* __restrict__ Y, const float* __restrict__ scb,
                           float* __restrict__ bsum, float* __restrict__ bsq, int N) {
    int t = threadIdx.x;
    int c0 = t * 2;
    float bb0 = scb[c0], bb1 = scb[c0 + 1];
    float s0 = 0.f, s1 = 0.f, q0 = 0.f, q1 = 0.f;
    for (int r = blockIdx.x; r < N; r += gridDim.x) {
        float2 v = *(const float2*)(Y + (size_t)r * 1024 + 512 + c0);
        float a0 = v.x + bb0, a1 = v.y + bb1;
        s0 += a0; s1 += a1; q0 += a0 * a0; q1 += a1 * a1;
    }
    atomicAdd(&bsum[c0], s0);     atomicAdd(&bsum[c0 + 1], s1);
    atomicAdd(&bsq[c0], q0);      atomicAdd(&bsq[c0 + 1], q1);
}

__global__ void k_bn_finalize(const float* __restrict__ bsum, const float* __restrict__ bsq,
                              float* __restrict__ bm, float* __restrict__ br, int N) {
    int c = blockIdx.x * blockDim.x + threadIdx.x;
    if (c >= 512) return;
    float inv = 1.f / (float)N;
    float mu = bsum[c] * inv;
    float var = bsq[c] * inv - mu * mu;
    bm[c] = mu;
    br[c] = rsqrtf(var + BN_EPS);
}

// ---------------------------------------------------------------- fused LN(y1)+BN(y2)+leaky -> h (bf16), 128 thr/row
__global__ void k_fuse_ln_bn(const float* __restrict__ Y,
                             const float* __restrict__ b1,  const float* __restrict__ lng,
                             const float* __restrict__ lnb, const float* __restrict__ scb,
                             const float* __restrict__ bm,  const float* __restrict__ br,
                             const float* __restrict__ bng, const float* __restrict__ bnb,
                             u16* __restrict__ H) {
    __shared__ float red[256];
    const int row = blockIdx.x, t = threadIdx.x;
    const float* y = Y + (size_t)row * 1024;
    const int c = t * 4;

    float4 f  = *(const float4*)(y + c);
    float4 bb = *(const float4*)(b1 + c);
    f.x += bb.x; f.y += bb.y; f.z += bb.z; f.w += bb.w;
    float s = f.x + f.y + f.z + f.w;
    float q = f.x * f.x + f.y * f.y + f.z * f.z + f.w * f.w;
    red[t] = s; red[128 + t] = q;
    __syncthreads();
    for (int o = 64; o > 0; o >>= 1) {
        if (t < o) { red[t] += red[t + o]; red[128 + t] += red[128 + t + o]; }
        __syncthreads();
    }
    const float mean = red[0] * (1.f / 512.f);
    const float var  = red[128] * (1.f / 512.f) - mean * mean;
    const float rinv = rsqrtf(var + LN_EPS);

    float4 g4 = *(const float4*)(lng + c);
    float4 l4 = *(const float4*)(lnb + c);
    float4 y2 = *(const float4*)(y + 512 + c);
    float4 sb = *(const float4*)(scb + c);
    float4 m4 = *(const float4*)(bm + c);
    float4 r4 = *(const float4*)(br + c);
    float4 G4 = *(const float4*)(bng + c);
    float4 B4 = *(const float4*)(bnb + c);

    float h0 = leaky((f.x - mean) * rinv * g4.x + l4.x + (y2.x + sb.x - m4.x) * r4.x * G4.x + B4.x);
    float h1 = leaky((f.y - mean) * rinv * g4.y + l4.y + (y2.y + sb.y - m4.y) * r4.y * G4.y + B4.y);
    float h2 = leaky((f.z - mean) * rinv * g4.z + l4.z + (y2.z + sb.z - m4.z) * r4.z * G4.z + B4.z);
    float h3 = leaky((f.w - mean) * rinv * g4.w + l4.w + (y2.w + sb.w - m4.w) * r4.w * G4.w + B4.w);

    uint2 o;
    o.x = (unsigned)f2b(h0) | ((unsigned)f2b(h1) << 16);
    o.y = (unsigned)f2b(h2) | ((unsigned)f2b(h3) << 16);
    *(uint2*)(H + (size_t)row * 512 + c) = o;
}

// ---------------------------------------------------------------- degree / dinv
__global__ void k_deg_init(float* deg, int N) {
    int i = blockIdx.x * blockDim.x + threadIdx.x;
    if (i < N) deg[i] = 1.f;                     // self loop
}
__global__ void k_deg_edges(const int* __restrict__ ei, float* __restrict__ deg, int E) {
    int e = blockIdx.x * blockDim.x + threadIdx.x;
    if (e < E) atomicAdd(&deg[ei[E + e]], 1.f);  // targets = ei[1]
}
__global__ void k_deg_rsqrt(const float* __restrict__ deg, float* __restrict__ dinv, int N) {
    int i = blockIdx.x * blockDim.x + threadIdx.x;
    if (i < N) dinv[i] = rsqrtf(deg[i]);
}

// ---------------------------------------------------------------- self-loop init: O = XW * dinv^2 (plain writes)
__global__ void k_selfloop(const float* __restrict__ XW, const float* __restrict__ dinv,
                           float* __restrict__ O, int N) {
    int tid = blockIdx.x * blockDim.x + threadIdx.x;
    int row = tid >> 6;
    if (row >= N) return;
    int cs = (tid & 63) * 4;
    float w = dinv[row]; w *= w;
    float4 v = *(const float4*)(XW + (size_t)row * 256 + cs);
    float4 o; o.x = v.x * w; o.y = v.y * w; o.z = v.z * w; o.w = v.w * w;
    *(float4*)(O + (size_t)row * 256 + cs) = o;
}

// ---------------------------------------------------------------- edge scatter-add (64 lanes per edge)
__global__ void k_scatter(const float* __restrict__ XW, const int* __restrict__ ei,
                          const float* __restrict__ dinv, float* __restrict__ O, int E) {
    int tid = blockIdx.x * blockDim.x + threadIdx.x;
    int e = tid >> 6;
    if (e >= E) return;
    int cs = (tid & 63) * 4;
    int r = ei[e], c = ei[E + e];
    float w = dinv[r] * dinv[c];
    float4 v = *(const float4*)(XW + (size_t)r * 256 + cs);
    float* dst = O + (size_t)c * 256 + cs;
    atomicAdd(dst + 0, v.x * w);
    atomicAdd(dst + 1, v.y * w);
    atomicAdd(dst + 2, v.z * w);
    atomicAdd(dst + 3, v.w * w);
}

// ---------------------------------------------------------------- final LayerNorm + leaky, in place on d_out (64 thr/row)
__global__ void k_final_ln(float* __restrict__ O, const float* __restrict__ gb,
                           const float* __restrict__ g, const float* __restrict__ b) {
    __shared__ float red[128];
    const int row = blockIdx.x, t = threadIdx.x;
    float* o = O + (size_t)row * 256;
    const int c = t * 4;
    float4 v  = *(const float4*)(o + c);
    float4 bb = *(const float4*)(gb + c);
    v.x += bb.x; v.y += bb.y; v.z += bb.z; v.w += bb.w;
    float s = v.x + v.y + v.z + v.w;
    float q = v.x * v.x + v.y * v.y + v.z * v.z + v.w * v.w;
    red[t] = s; red[64 + t] = q;
    __syncthreads();
    for (int off = 32; off > 0; off >>= 1) {
        if (t < off) { red[t] += red[t + off]; red[64 + t] += red[64 + t + off]; }
        __syncthreads();
    }
    const float mean = red[0] * (1.f / 256.f);
    const float rinv = rsqrtf(red[64] * (1.f / 256.f) - mean * mean + LN_EPS);
    float4 g4 = *(const float4*)(g + c);
    float4 b4 = *(const float4*)(b + c);
    float4 r;
    r.x = leaky((v.x - mean) * rinv * g4.x + b4.x);
    r.y = leaky((v.y - mean) * rinv * g4.y + b4.y);
    r.z = leaky((v.z - mean) * rinv * g4.z + b4.z);
    r.w = leaky((v.w - mean) * rinv * g4.w + b4.w);
    *(float4*)(o + c) = r;
}

// ================================================================ host launcher
extern "C" void kernel_launch(void* const* d_in, const int* in_sizes, int n_in,
                              void* d_out, int out_size, void* d_ws, size_t ws_size,
                              hipStream_t stream) {
    const float* x    = (const float*)d_in[0];
    const int*   ei   = (const int*)d_in[1];
    const float* W1   = (const float*)d_in[2];
    const float* b1   = (const float*)d_in[3];
    const float* ln1g = (const float*)d_in[4];
    const float* ln1b = (const float*)d_in[5];
    const float* scW  = (const float*)d_in[6];
    const float* scb  = (const float*)d_in[7];
    const float* bng  = (const float*)d_in[8];
    const float* bnb  = (const float*)d_in[9];
    const float* gW   = (const float*)d_in[10];
    const float* gb   = (const float*)d_in[11];
    const float* ln2g = (const float*)d_in[12];
    const float* ln2b = (const float*)d_in[13];
    float* out = (float*)d_out;

    const int N = in_sizes[0] / 1024;   // 50000
    const int E = in_sizes[1] / 2;      // 800000

    // workspace carve-out (256B aligned)
    char* ws = (char*)d_ws;
    size_t off = 0;
    auto carve = [&](size_t bytes) {
        void* p = ws + off;
        off = (off + bytes + 255) & ~(size_t)255;
        return p;
    };
    u16*   xb    = (u16*)carve((size_t)N * 1024 * 2);
    u16*   wcat  = (u16*)carve((size_t)1024 * 1024 * 2);   // rows 0..511 = W1, 512..1023 = scW
    u16*   gw    = (u16*)carve((size_t)256 * 512 * 2);
    float* Y     = (float*)carve((size_t)N * 1024 * 4);
    u16*   hb    = (u16*)carve((size_t)N * 512 * 2);
    float* xw    = (float*)carve((size_t)N * 256 * 4);
    float* bnsum = (float*)carve(1024 * 4);                // [sum|sqsum]
    float* bnsq  = bnsum + 512;
    float* bm    = (float*)carve(512 * 4);
    float* br    = (float*)carve(512 * 4);
    float* deg   = (float*)carve((size_t)N * 4);
    float* dinv  = (float*)carve((size_t)N * 4);
    (void)ws_size; (void)n_in; (void)out_size;

    auto blk8 = [](int n) { return (n / 8 + 255) / 256; };

    // 1) precision conversion (x streamed once; weights fused W1||scW)
    k_f32_to_bf16<<<blk8(N * 1024), 256, 0, stream>>>(x, xb, N * 1024 / 8);
    k_f32_to_bf16<<<blk8(512 * 1024), 256, 0, stream>>>(W1, wcat, 512 * 1024 / 8);
    k_f32_to_bf16<<<blk8(512 * 1024), 256, 0, stream>>>(scW, wcat + 512 * 1024, 512 * 1024 / 8);
    k_f32_to_bf16<<<blk8(256 * 512), 256, 0, stream>>>(gW, gw, 256 * 512 / 8);

    // 2) fused dual GEMM: Y[:, :512] = x@W1^T, Y[:, 512:] = x@scW^T
    k_gemm_bf16_wmma<1024, 8, 8><<<N / 16, 256, 0, stream>>>(xb, wcat, Y);

    // 3) BatchNorm stats + LN/BN/leaky fusion -> h (bf16)
    k_zero_f32<<<(1024 + 255) / 256, 256, 0, stream>>>(bnsum, 1024);
    k_bn_stats<<<256, 256, 0, stream>>>(Y, scb, bnsum, bnsq, N);
    k_bn_finalize<<<2, 256, 0, stream>>>(bnsum, bnsq, bm, br, N);
    k_fuse_ln_bn<<<N, 128, 0, stream>>>(Y, b1, ln1g, ln1b, scb, bm, br, bng, bnb, hb);

    // 4) GEMM: xw = h @ gcn_W^T
    k_gemm_bf16_wmma<512, 4, 4><<<N / 16, 128, 0, stream>>>(hb, gw, xw);

    // 5) GCN degree norm + scatter
    k_deg_init<<<(N + 255) / 256, 256, 0, stream>>>(deg, N);
    k_deg_edges<<<(E + 255) / 256, 256, 0, stream>>>(ei, deg, E);
    k_deg_rsqrt<<<(N + 255) / 256, 256, 0, stream>>>(deg, dinv, N);
    k_selfloop<<<(N * 64 + 255) / 256, 256, 0, stream>>>(xw, dinv, out, N);
    k_scatter<<<(E * 64 + 255) / 256, 256, 0, stream>>>(xw, ei, dinv, out, E);

    // 6) final LayerNorm + leaky, in place
    k_final_ln<<<N, 64, 0, stream>>>(out, gb, ln2g, ln2b);
}
